// BlockS_11751030522613
// MI455X (gfx1250) — compile-verified, matrix operand order checked by reading, not run on hardware
//
#include <hip/hip_runtime.h>

// ---------------------------------------------------------------------------
// ST-GCN block forward for MI455X (gfx1250), wave32 + v_wmma_f32_16x16x32_f16
// All WMMA operands are stored so each lane reads two contiguous 16B chunks
// (A: [row][k] row-major; B: stored transposed [col][k] row-major).
// Fusion GEMM stages its block-shared weight tile in LDS via
// GLOBAL_LOAD_ASYNC_TO_LDS_B128 and feeds WMMA A from ds_load_b128.
// ---------------------------------------------------------------------------

typedef __attribute__((ext_vector_type(16))) _Float16 v16h;
typedef __attribute__((ext_vector_type(8)))  _Float16 v8h;
typedef __attribute__((ext_vector_type(8)))  float    v8f;
typedef __attribute__((ext_vector_type(4)))  float    v4f;
typedef __attribute__((ext_vector_type(4)))  int      v4i;

#if defined(__HIP_DEVICE_COMPILE__)
typedef __attribute__((address_space(1))) v4i gv4i;   // global int4
typedef __attribute__((address_space(3))) v4i lv4i;   // LDS int4
#endif

#define WMMA16(a, b, c) \
  __builtin_amdgcn_wmma_f32_16x16x32_f16(false, (a), false, (b), (short)0, (c), false, false)

// problem sizes
constexpr int kN   = 32;
constexpr int kC   = 64;
constexpr int kT   = 256;
constexpr int kV   = 25;
constexpr int kBC  = 64;
constexpr int kOUT = 256;
constexpr int kPOS = kT * kV;     // 6400
constexpr int kPT  = kBC * kT;    // 16384 gram reduction length
constexpr int kNTP = kPOS / 16;   // 400
constexpr int kNPERC = 409600;    // 64*6400 per-n stride (xt/e1h/e2t, per-(i,n) xat/ypre)
constexpr int kLDA = 264;         // padded LDS row stride (halves): 4-bank row skew

// workspace layout (bytes)
constexpr size_t AC_OFF  = 0;                      // Ac[4][25][25] f32
constexpr size_t AFT_OFF = 16384;                  // Af^T padded [4][32][32][32] f16
constexpr size_t CA_OFF  = AFT_OFF + 262144;       // CA accum [32][25][25] f32
constexpr size_t ST_OFF  = CA_OFF + 81920;         // stats 1024 f32
constexpr size_t BN_OFF  = ST_OFF + 8192;          // bn params 1024 f32
constexpr size_t XT_OFF  = BN_OFF + 8192;          // xt[n][pos][c] f16      (26,214,400)
constexpr size_t XHP_OFF = XT_OFF + 26214400;      // xhp[n][ct][32] f16     (33,554,432)
constexpr size_t R1_OFF  = XHP_OFF + 33554432;     // e1h|e2t -> xat -> catt (104,857,600)
constexpr size_t R2_OFF  = R1_OFF + 104857600;     // ypre f16               (104,857,600)

// ---------------------------------------------------------------------------
// Vectorized WMMA operand loaders.
// Lane layout (ISA 7.12.2): lane l, hi=l>>4, idx=l&15; half j maps to
// K = k0 + (j&7) + (j&8?16:0) + hi*8  -> two contiguous 8-half (16B) chunks.
// ---------------------------------------------------------------------------

// f16 operand, [row][k] row-major, no guards (dims exact, 16B aligned)
__device__ __forceinline__ v16h ldT_h(const _Float16* __restrict__ base, int row0,
                                      int ld, int k0) {
  int l = threadIdx.x & 31, hi = l >> 4;
  const _Float16* p = base + (size_t)(row0 + (l & 15)) * ld + k0 + hi * 8;
  v8h c0 = *(const v8h*)(p);
  v8h c1 = *(const v8h*)(p + 16);
  return __builtin_shufflevector(c0, c1, 0, 1, 2, 3, 4, 5, 6, 7,
                                 8, 9, 10, 11, 12, 13, 14, 15);
}

// f16 operand with row guard (clamped address + branchless select)
__device__ __forceinline__ v16h ldT_hg(const _Float16* __restrict__ base, int row0,
                                       int ld, int k0, int rmax) {
  int l = threadIdx.x & 31, hi = l >> 4;
  int row = row0 + (l & 15);
  bool rv = row < rmax;
  int rc = rv ? row : (rmax - 1);
  const _Float16* p = base + (size_t)rc * ld + k0 + hi * 8;
  v8h z = {};
  v8h c0 = *(const v8h*)(p);
  v8h c1 = *(const v8h*)(p + 16);
  c0 = rv ? c0 : z;
  c1 = rv ? c1 : z;
  return __builtin_shufflevector(c0, c1, 0, 1, 2, 3, 4, 5, 6, 7,
                                 8, 9, 10, 11, 12, 13, 14, 15);
}

// f32 weights, [row][k] row-major, converted to f16 in-register
__device__ __forceinline__ v16h ldT_f(const float* __restrict__ base, int row0,
                                      int ld, int k0) {
  int l = threadIdx.x & 31, hi = l >> 4;
  const float* p = base + (size_t)(row0 + (l & 15)) * ld + k0 + hi * 8;
  v4f f0 = *(const v4f*)(p);
  v4f f1 = *(const v4f*)(p + 4);
  v4f f2 = *(const v4f*)(p + 16);
  v4f f3 = *(const v4f*)(p + 20);
  v16h a;
#pragma unroll
  for (int j = 0; j < 4; ++j) {
    a[j]      = (_Float16)f0[j];
    a[4 + j]  = (_Float16)f1[j];
    a[8 + j]  = (_Float16)f2[j];
    a[12 + j] = (_Float16)f3[j];
  }
  return a;
}

// ---------------------------------------------------------------------------
// Kernels
// ---------------------------------------------------------------------------

__global__ void k_zero(float* p, int n) {
  int i = blockIdx.x * blockDim.x + threadIdx.x;
  if (i < n) p[i] = 0.0f;
}

// x f32 -> f16 twice: xhp[n][ct][32] zero-padded (graph-A) and xt[n][pos][c] (proj-B)
__global__ void k_cvt(const float* __restrict__ x, _Float16* __restrict__ xhp,
                      _Float16* __restrict__ xt) {
  int idx = blockIdx.x * blockDim.x + threadIdx.x;   // over [n][ct][32]
  int vp = idx & 31;
  int ct = (idx >> 5) & 16383;
  int n  = idx >> 19;
  _Float16 h = (_Float16)0.0f;
  if (vp < kV) {
    h = (_Float16)x[n * kNPERC + ct * kV + vp];
    int c = ct >> 8, t = ct & 255;
    int pos = t * kV + vp;
    xt[((size_t)n * kPOS + pos) * kC + c] = h;
  }
  xhp[idx] = h;
}

// constant adjacency stack Ac[4][25][25]
__global__ void k_ac(const float* __restrict__ Acst, float* __restrict__ Ac) {
  int idx = blockIdx.x * blockDim.x + threadIdx.x;
  if (idx >= kV * kV) return;
  int v = idx / kV, w = idx % kV;
  const float* A0 = Acst;
  const float* A1 = Acst + kV * kV;
  const float* A2 = Acst + 2 * kV * kV;
  float a3s = A1[idx] + A2[idx];
  float I = (v == w) ? 1.0f : 0.0f;
  Ac[idx]               = A0[idx];
  Ac[kV * kV + idx]     = A0[idx] - A1[idx];
  Ac[2 * kV * kV + idx] = 4.0f * a3s * a3s - a3s - 2.0f * I;
  float mx = -3.4e38f, xv = 0.0f;
  for (int r = 0; r < kV; ++r) {
    float a = A1[r * kV + w] + A2[r * kV + w];
    float a2 = a * a;
    float m4 = 8.0f * a2 * a2 - 4.0f * a2 - 4.0f * a + ((r == w) ? 1.0f : 0.0f);
    float xs = m4 / (float)kV;
    if (r == v) xv = xs;
    mx = fmaxf(mx, xs);
  }
  float s = 0.0f;
  for (int r = 0; r < kV; ++r) {
    float a = A1[r * kV + w] + A2[r * kV + w];
    float a2 = a * a;
    float m4 = 8.0f * a2 * a2 - 4.0f * a2 - 4.0f * a + ((r == w) ? 1.0f : 0.0f);
    s += expf(m4 / (float)kV - mx);
  }
  Ac[3 * kV * kV + idx] = expf(xv - mx) / s;
}

// e1/e2 projections: D[p][pos] = sum_c w[p][c]*x[n][pos][c] + bias,
// scatter into gram layouts e1h[n][v][pt], e2t[n][w][pt] (both [row][pt])
__global__ void k_eproj(const _Float16* __restrict__ xt,
                        const float* __restrict__ p1w, const float* __restrict__ p1b,
                        const float* __restrict__ p2w, const float* __restrict__ p2b,
                        _Float16* __restrict__ e1h, _Float16* __restrict__ e2t) {
  int wid = blockIdx.x * (blockDim.x >> 5) + (threadIdx.x >> 5);
  int nt = wid % kNTP; wid /= kNTP;
  int mt = wid % 4;    wid /= 4;
  int n  = wid % kN;   wid /= kN;
  int proj = wid;
  if (proj >= 2) return;
  const float* w  = proj ? p2w : p1w;
  const float* bs = proj ? p2b : p1b;
  const _Float16* B = xt + (size_t)n * kNPERC;
  v8f c = {};
#pragma unroll
  for (int ks = 0; ks < 2; ++ks) {
    v16h a = ldT_f(w, mt * 16, kC, ks * 32);
    v16h b = ldT_h(B, nt * 16, kC, ks * 32);
    c = WMMA16(a, b, c);
  }
  int l = threadIdx.x & 31;
  int pos = nt * 16 + (l & 15);
  int rb  = mt * 16 + ((l >> 4) << 3);
  int tt = pos / kV, vv = pos % kV;
  _Float16* e = proj ? e2t : e1h;
#pragma unroll
  for (int r = 0; r < 8; ++r) {
    int p = rb + r;
    float val = c[r] + bs[p];
    e[(size_t)n * kNPERC + vv * kPT + p * kT + tt] = (_Float16)val;
  }
}

// gram, split-K: CA[n][v][w] += sum_{k in slice} e1h[n][v][k]*e2t[n][w][k]
__global__ void k_gram(const _Float16* __restrict__ e1h,
                       const _Float16* __restrict__ e2t,
                       float* __restrict__ CA) {
  int wid = blockIdx.x * (blockDim.x >> 5) + (threadIdx.x >> 5);
  int slice = wid & 31;  wid >>= 5;
  int ntile = wid & 1;
  int mtile = (wid >> 1) & 1;
  int n = wid >> 2;
  if (n >= kN) return;
  const _Float16* A = e1h + (size_t)n * kNPERC;
  const _Float16* B = e2t + (size_t)n * kNPERC;
  int kbeg = slice * 512;
  v8f c = {};
#pragma unroll
  for (int ks = 0; ks < 16; ++ks) {
    int k0 = kbeg + ks * 32;
    __builtin_prefetch(A + (size_t)(mtile * 16) * kPT + k0 + 256, 0, 1);
    __builtin_prefetch(B + (size_t)(ntile * 16) * kPT + k0 + 256, 0, 1);
    v16h a = ldT_hg(A, mtile * 16, kPT, k0, kV);
    v16h b = ldT_hg(B, ntile * 16, kPT, k0, kV);
    c = WMMA16(a, b, c);
  }
  int l = threadIdx.x & 31;
  int w  = ntile * 16 + (l & 15);
  int rb = mtile * 16 + ((l >> 4) << 3);
#pragma unroll
  for (int r = 0; r < 8; ++r) {
    int v = rb + r;
    if (v < kV && w < kV) atomicAdd(&CA[n * kV * kV + v * kV + w], c[r]);
  }
}

// softmax(CA*sc, axis=-2) + build padded transposed adjacency AfT[i][n][w][v]
__global__ void k_af(const float* __restrict__ CA, const float* __restrict__ Ac,
                     const float* __restrict__ PA, const float* __restrict__ PCA,
                     _Float16* __restrict__ AfT) {
  int idx = blockIdx.x * blockDim.x + threadIdx.x;
  if (idx >= 4 * kN * 32 * 32) return;
  int vv = idx & 31, ww = (idx >> 5) & 31;     // AfT[i][n][ww][vv]
  int n = (idx >> 10) & 31, i = idx >> 15;
  const float sc = 1.0f / (float)(kBC * kT);
  float val = 0.0f;
  if (vv < kV && ww < kV) {
    float mx = -3.4e38f;
    for (int r = 0; r < kV; ++r) mx = fmaxf(mx, CA[n * kV * kV + r * kV + ww] * sc);
    float s = 0.0f;
    for (int r = 0; r < kV; ++r) s += expf(CA[n * kV * kV + r * kV + ww] * sc - mx);
    float smx = expf(CA[n * kV * kV + vv * kV + ww] * sc - mx) / s;
    val = Ac[i * kV * kV + vv * kV + ww] + PCA[0] * smx + PA[i * kV * kV + vv * kV + ww];
  }
  AfT[((size_t)(i * kN + n) * 32 + ww) * 32 + vv] = (_Float16)val;
}

// graph matmul: xat[i][n][pos=(t,w)][c] = sum_v xhp[n][(c,t)][v] * AfT[i][n][w][v]
__global__ void k_graph(const _Float16* __restrict__ xhp,
                        const _Float16* __restrict__ AfT,
                        _Float16* __restrict__ xat) {
  int wid = blockIdx.x * (blockDim.x >> 5) + (threadIdx.x >> 5);
  int ntile = wid & 1;   wid >>= 1;
  int mt = wid & 1023;   wid >>= 10;
  int n = wid & 31;
  int i = wid >> 5;
  if (i >= 4) return;
  const _Float16* A = xhp + (size_t)n * (kPT * 32);
  const _Float16* B = AfT + ((size_t)(i * kN + n) << 10);
  v16h a = ldT_h(A, mt * 16, 32, 0);
  v16h b = ldT_h(B, ntile * 16, 32, 0);
  v8f c = {};
  c = WMMA16(a, b, c);
  int l = threadIdx.x & 31;
  int w  = ntile * 16 + (l & 15);
  int rb = mt * 16 + ((l >> 4) << 3);
  if (w < kV) {
#pragma unroll
    for (int r = 0; r < 8; ++r) {
      int R = rb + r;               // R = c*T + t
      int cc = R >> 8, t = R & 255;
      int pos = t * kV + w;
      xat[((size_t)(i * kN + n) * kPOS + pos) * kC + cc] = (_Float16)c[r];
    }
  }
}

// branch conv: y[i][n][o][pos] = sum_c bw[i][o][c]*xat[i][n][pos][c] + BN stats
__global__ void k_bconv(const float* __restrict__ bw,
                        const _Float16* __restrict__ xat,
                        _Float16* __restrict__ ypre,
                        float* __restrict__ stats) {
  int wid = blockIdx.x * (blockDim.x >> 5) + (threadIdx.x >> 5);
  int nt = wid % kNTP; wid /= kNTP;
  int mt = wid % 4;    wid /= 4;
  int n  = wid % kN;   wid /= kN;
  int i = wid;
  if (i >= 4) return;
  const float* A = bw + i * kBC * kC;
  const _Float16* B = xat + (size_t)(i * kN + n) * kNPERC;
  v8f c = {};
#pragma unroll
  for (int ks = 0; ks < 2; ++ks) {
    v16h a = ldT_f(A, mt * 16, kC, ks * 32);
    v16h b = ldT_h(B, nt * 16, kC, ks * 32);
    c = WMMA16(a, b, c);
  }
  int l = threadIdx.x & 31;
  int pos = nt * 16 + (l & 15);
  int rb  = mt * 16 + ((l >> 4) << 3);
#pragma unroll
  for (int r = 0; r < 8; ++r) {
    int o = rb + r;
    float val = c[r];
    ypre[(size_t)(i * kN + n) * kNPERC + o * kPOS + pos] = (_Float16)val;
    float s = val, q = val * val;
    s += __shfl_xor(s, 1); s += __shfl_xor(s, 2);
    s += __shfl_xor(s, 4); s += __shfl_xor(s, 8);
    q += __shfl_xor(q, 1); q += __shfl_xor(q, 2);
    q += __shfl_xor(q, 4); q += __shfl_xor(q, 8);
    if ((l & 15) == 0) {
      atomicAdd(&stats[i * kBC + o], s);
      atomicAdd(&stats[256 + i * kBC + o], q);
    }
  }
}

__global__ void k_finbn(const float* __restrict__ stats,
                        const float* __restrict__ gamma,
                        const float* __restrict__ beta,
                        float* __restrict__ bnp, int nch, float inv_count) {
  int o = blockIdx.x * blockDim.x + threadIdx.x;
  if (o >= nch) return;
  float mean = stats[o] * inv_count;
  float var = stats[nch + o] * inv_count - mean * mean;
  float sc = gamma[o] * rsqrtf(var + 1e-5f);
  bnp[o] = sc;
  bnp[nch + o] = beta[o] - mean * sc;
}

// apply branch BN + ReLU, pack transposed cat layout catt[n][pos][oc] f16
__global__ void k_apply_relu(const _Float16* __restrict__ ypre,
                             const float* __restrict__ bnp,
                             _Float16* __restrict__ catt, int total) {
  int idx = blockIdx.x * blockDim.x + threadIdx.x;
  if (idx >= total) return;
  int pos = idx % kPOS;
  int tmp = idx / kPOS;
  int o = tmp % kBC; tmp /= kBC;
  int n = tmp % kN;
  int i = tmp / kN;
  int ch = i * kBC + o;
  float val = (float)ypre[idx] * bnp[ch] + bnp[256 + ch];
  val = fmaxf(val, 0.0f);
  catt[((size_t)n * kPOS + pos) * kOUT + ch] = (_Float16)val;
}

// fusion conv: out[n][O][pos] = sum_oc fw[O][oc]*catt[n][pos][oc] + BN stats.
// All 4 waves of a block share the same 16-row fw tile -> stage it in LDS
// once (async-to-LDS when available), convert to f16, feed WMMA A from LDS.
__global__ void k_fusion(const float* __restrict__ fw,
                         const _Float16* __restrict__ catt,
                         float* __restrict__ out,
                         float* __restrict__ stats) {
  __shared__ float    lAf[16 * kOUT];        // raw f32 staging (16 KB)
  __shared__ _Float16 lA[16 * kLDA];         // f16 A tile, padded rows (8.25 KB)

  int wid = blockIdx.x * (blockDim.x >> 5) + (threadIdx.x >> 5);
  int nt = wid % kNTP; wid /= kNTP;
  int mt = wid % 16;   wid /= 16;
  int n = wid;

  // block-uniform weight tile (all waves share mt)
  const float* Ag = fw + (size_t)mt * 16 * kOUT;
#if defined(__HIP_DEVICE_COMPILE__) && \
    __has_builtin(__builtin_amdgcn_global_load_async_to_lds_b128)
  for (int e = threadIdx.x; e < 16 * kOUT / 4; e += 128) {   // 1024 x 16B chunks
    float* gsrc = const_cast<float*>(Ag + e * 4);
    __builtin_amdgcn_global_load_async_to_lds_b128(
        (gv4i*)gsrc, (lv4i*)(lAf + e * 4), 0, 0);
  }
  asm volatile("s_wait_asynccnt 0x0" ::: "memory");
  __syncthreads();
  for (int e = threadIdx.x; e < 16 * kOUT; e += 128) {
    lA[(e >> 8) * kLDA + (e & 255)] = (_Float16)lAf[e];
  }
#else
  for (int e = threadIdx.x; e < 16 * kOUT; e += 128) {
    lA[(e >> 8) * kLDA + (e & 255)] = (_Float16)Ag[e];
  }
  (void)lAf;
#endif
  __syncthreads();

  const _Float16* B = catt + (size_t)n * kOUT * kPOS;
  v8f c = {};
#pragma unroll
  for (int ks = 0; ks < 8; ++ks) {
    __builtin_prefetch(B + (size_t)(nt * 16) * kOUT + ks * 32 + 64, 0, 1);
    v16h a = ldT_h(lA, 0, kLDA, ks * 32);     // ds_load_b128 from LDS tile
    v16h b = ldT_h(B, nt * 16, kOUT, ks * 32);
    c = WMMA16(a, b, c);
  }
  int l = threadIdx.x & 31;
  int pos = nt * 16 + (l & 15);
  int rb  = mt * 16 + ((l >> 4) << 3);
#pragma unroll
  for (int r = 0; r < 8; ++r) {
    int O = rb + r;
    float val = c[r];
    out[((size_t)n * kOUT + O) * kPOS + pos] = val;
    float s = val, q = val * val;
    s += __shfl_xor(s, 1); s += __shfl_xor(s, 2);
    s += __shfl_xor(s, 4); s += __shfl_xor(s, 8);
    q += __shfl_xor(q, 1); q += __shfl_xor(q, 2);
    q += __shfl_xor(q, 4); q += __shfl_xor(q, 8);
    if ((l & 15) == 0) {
      atomicAdd(&stats[512 + O], s);
      atomicAdd(&stats[768 + O], q);
    }
  }
}

__global__ void k_apply_out(float* __restrict__ out, const float* __restrict__ bnp,
                            int total) {
  int idx = blockIdx.x * blockDim.x + threadIdx.x;
  if (idx >= total) return;
  int O = (idx / kPOS) % kOUT;
  out[idx] = out[idx] * bnp[512 + O] + bnp[768 + O];
}

// ---------------------------------------------------------------------------
extern "C" void kernel_launch(void* const* d_in, const int* in_sizes, int n_in,
                              void* d_out, int out_size, void* d_ws, size_t ws_size,
                              hipStream_t stream) {
  const float* x    = (const float*)d_in[0];
  const float* Acst = (const float*)d_in[1];
  const float* PA   = (const float*)d_in[2];
  const float* PCA  = (const float*)d_in[3];
  const float* p1w  = (const float*)d_in[4];
  const float* p1b  = (const float*)d_in[5];
  const float* p2w  = (const float*)d_in[6];
  const float* p2b  = (const float*)d_in[7];
  const float* bw   = (const float*)d_in[8];
  const float* bgam = (const float*)d_in[9];
  const float* bbet = (const float*)d_in[10];
  const float* fw   = (const float*)d_in[11];
  const float* fgam = (const float*)d_in[12];
  const float* fbet = (const float*)d_in[13];
  float* out = (float*)d_out;

  char* ws = (char*)d_ws;
  float*    Ac    = (float*)(ws + AC_OFF);
  _Float16* AfT   = (_Float16*)(ws + AFT_OFF);
  float*    CA    = (float*)(ws + CA_OFF);
  float*    stats = (float*)(ws + ST_OFF);
  float*    bnp   = (float*)(ws + BN_OFF);
  _Float16* xt    = (_Float16*)(ws + XT_OFF);
  _Float16* xhp   = (_Float16*)(ws + XHP_OFF);
  _Float16* e1h   = (_Float16*)(ws + R1_OFF);
  _Float16* e2t   = (_Float16*)(ws + R1_OFF + (size_t)kN * kNPERC * 2);
  _Float16* xat   = (_Float16*)(ws + R1_OFF);   // reuses e1h/e2t after gram
  _Float16* catt  = (_Float16*)(ws + R1_OFF);   // reuses xat after branch conv
  _Float16* ypre  = (_Float16*)(ws + R2_OFF);

  const int total_y = 4 * kN * kBC * kPOS;        // 52428800

  // 0) zero BN stats and CA accumulators (ws is poisoned by harness)
  k_zero<<<4, 256, 0, stream>>>(stats, 1024);
  k_zero<<<80, 256, 0, stream>>>(CA, kN * kV * kV);
  // 1) constant adjacencies
  k_ac<<<3, 256, 0, stream>>>(Acst, Ac);
  // 2) x -> f16 (padded graph-A copy + channel-last proj-B copy)
  k_cvt<<<65536, 256, 0, stream>>>(x, xhp, xt);
  // 3) e1/e2 projections (102400 waves)
  k_eproj<<<25600, 128, 0, stream>>>(xt, p1w, p1b, p2w, p2b, e1h, e2t);
  // 4) gram, split-K (4096 waves)
  k_gram<<<1024, 128, 0, stream>>>(e1h, e2t, CA);
  // 5) softmax + adjacency build (transposed, padded)
  k_af<<<512, 256, 0, stream>>>(CA, Ac, PA, PCA, AfT);
  // 6) graph matmul (262144 waves)
  k_graph<<<65536, 128, 0, stream>>>(xhp, AfT, xat);
  // 7) branch conv + stats (204800 waves)
  k_bconv<<<51200, 128, 0, stream>>>(bw, xat, ypre, stats);
  // 8) finalize branch BN
  k_finbn<<<1, 256, 0, stream>>>(stats, bgam, bbet, bnp, 256,
                                 1.0f / (float)(kN * kT * kV));
  // 9) apply BN + ReLU, pack transposed cat
  k_apply_relu<<<total_y / 256, 256, 0, stream>>>(ypre, bnp, catt, total_y);
  // 10) fusion conv (LDS-staged A via async copy) -> d_out + stats
  k_fusion<<<51200, 128, 0, stream>>>(fw, catt, out, stats);
  // 11) finalize fusion BN
  k_finbn<<<1, 256, 0, stream>>>(stats + 512, fgam, fbet, bnp + 512, 256,
                                 1.0f / (float)(kN * kT * kV));
  // 12) apply fusion BN in place
  k_apply_out<<<total_y / 256, 256, 0, stream>>>(out, bnp, total_y);

  (void)in_sizes; (void)n_in; (void)out_size; (void)ws_size;
}
